// SoftTreeEnsemble_72567767433903
// MI455X (gfx1250) — matrix operand order repaired and checked by reading, not compiled, not run on hardware
//
#include <hip/hip_runtime.h>
#include <hip/hip_bf16.h>

typedef __attribute__((ext_vector_type(16))) _Float16 v16h;
typedef __attribute__((ext_vector_type(8)))  _Float16 v8h;
typedef __attribute__((ext_vector_type(4)))  _Float16 v4h;
typedef __attribute__((ext_vector_type(8)))  float    v8f;

#define BATCH      4096
#define INPUT_DIM  1024
#define NUM_TREES  64
#define N_INTERNAL 63
#define N_LEAVES   64
#define LEAF_DIMS  10
#define ROW_TILE   32   // batch rows per workgroup (2 x 16-row WMMA subtiles)
#define T_HALF     32   // trees per workgroup (blockIdx.y selects half)

// ---------------------------------------------------------------------------
// Kernel A: f32 -> f16 conversion (vectorized x4)
// ---------------------------------------------------------------------------
__global__ __launch_bounds__(256) void cvt_f32_to_f16_x4(
    const float* __restrict__ src, _Float16* __restrict__ dst, int n4) {
  int i = blockIdx.x * blockDim.x + threadIdx.x;
  if (i < n4) {
    const float4 v = ((const float4*)src)[i];
    v4h o;
    o[0] = (_Float16)v.x;
    o[1] = (_Float16)v.y;
    o[2] = (_Float16)v.z;
    o[3] = (_Float16)v.w;
    ((v4h*)dst)[i] = o;
  }
}

// ---------------------------------------------------------------------------
// Kernel B: fused gate-GEMM (WMMA f16->f32, 2x2 register blocking) + sigmoid
// + tree cascade + leaf contraction. One WG = 32 batch rows x 32 trees.
// ---------------------------------------------------------------------------
__device__ __forceinline__ v16h combine16(v8h lo, v8h hi) {
  v16h r;
#pragma unroll
  for (int j = 0; j < 8; ++j) { r[j] = lo[j]; r[j + 8] = hi[j]; }
  return r;
}

__global__ __launch_bounds__(256) void soft_tree_main(
    const _Float16* __restrict__ xh,
    const _Float16* __restrict__ wh,
    const float*    __restrict__ bias,
    const float*    __restrict__ leaf_w,
    float*          __restrict__ part) {
  extern __shared__ char smem[];
  float* sh_g   = (float*)smem;                               // [63][32][32] gates
  float* sh_red = sh_g + N_INTERNAL * ROW_TILE * T_HALF;      // [32][32][10] partials

  const int tid   = threadIdx.x;
  const int lane  = tid & 31;
  const int wv    = tid >> 5;           // wave id, 0..7
  const int row0  = blockIdx.x * ROW_TILE;
  const int thalf = blockIdx.y;         // 0 or 1 -> trees [0,32) or [32,64)
  const int l15   = lane & 15;
  const int hi16  = lane >> 4;          // 0 for lanes 0-15, 1 for lanes 16-31

  // -------------------- GEMM phase: gates = sigmoid(x.W^T + b) ---------------
  // A fragment (16x32 f16): lane m holds K {ka..ka+7} and {ka+16..ka+23}, ka = hi16*8
  // B fragment (32x16 f16): lane n holds K {kb2..kb2+7} and {kb2+16..kb2+23}, kb2 = hi16*8
  const _Float16* arow0 = xh + (size_t)(row0 + l15) * INPUT_DIM;
  const _Float16* arow1 = arow0 + (size_t)16 * INPUT_DIM;
  const int ka  = hi16 ? 8  : 0;
  const int kb2 = hi16 ? 8 : 0;

  for (int n = wv; n < N_INTERNAL; n += 8) {
    const _Float16* b0 = wh + ((size_t)n * NUM_TREES + thalf * T_HALF + l15) * INPUT_DIM;
    const _Float16* b1 = b0 + (size_t)16 * INPUT_DIM;
    __builtin_prefetch(b0, 0, 3);
    __builtin_prefetch(b1, 0, 3);
    v8f acc00 = {};  // rows 0-15,  trees tg0
    v8f acc01 = {};  // rows 0-15,  trees tg1
    v8f acc10 = {};  // rows 16-31, trees tg0
    v8f acc11 = {};  // rows 16-31, trees tg1
#pragma unroll 4
    for (int k = 0; k < INPUT_DIM; k += 32) {
      v16h a0 = combine16(*(const v8h*)(arow0 + k + ka),
                          *(const v8h*)(arow0 + k + ka + 16));
      v16h a1 = combine16(*(const v8h*)(arow1 + k + ka),
                          *(const v8h*)(arow1 + k + ka + 16));
      v16h f0 = combine16(*(const v8h*)(b0 + k + kb2),
                          *(const v8h*)(b0 + k + kb2 + 16));
      v16h f1 = combine16(*(const v8h*)(b1 + k + kb2),
                          *(const v8h*)(b1 + k + kb2 + 16));
      acc00 = __builtin_amdgcn_wmma_f32_16x16x32_f16(false, a0, false, f0,
                                                     (short)0, acc00, false, false);
      acc01 = __builtin_amdgcn_wmma_f32_16x16x32_f16(false, a0, false, f1,
                                                     (short)0, acc01, false, false);
      acc10 = __builtin_amdgcn_wmma_f32_16x16x32_f16(false, a1, false, f0,
                                                     (short)0, acc10, false, false);
      acc11 = __builtin_amdgcn_wmma_f32_16x16x32_f16(false, a1, false, f1,
                                                     (short)0, acc11, false, false);
    }
    // bias depends only on (node, tree); C/D layout: vgpr r -> row r + hi16*8, col l15
    const float bv0 = bias[n * NUM_TREES + thalf * T_HALF + l15];
    const float bv1 = bias[n * NUM_TREES + thalf * T_HALF + 16 + l15];
#pragma unroll
    for (int r = 0; r < 8; ++r) {
      const int rowa = r + hi16 * 8;
      const float g00 = 1.0f / (1.0f + __expf(-(acc00[r] + bv0)));
      const float g01 = 1.0f / (1.0f + __expf(-(acc01[r] + bv1)));
      const float g10 = 1.0f / (1.0f + __expf(-(acc10[r] + bv0)));
      const float g11 = 1.0f / (1.0f + __expf(-(acc11[r] + bv1)));
      sh_g[(n * ROW_TILE + rowa) * T_HALF + l15]           = g00;
      sh_g[(n * ROW_TILE + rowa) * T_HALF + 16 + l15]      = g01;
      sh_g[(n * ROW_TILE + 16 + rowa) * T_HALF + l15]      = g10;
      sh_g[(n * ROW_TILE + 16 + rowa) * T_HALF + 16 + l15] = g11;
    }
  }
  __syncthreads();

  // -------------------- tree cascade + leaf contraction ----------------------
  for (int p = tid; p < ROW_TILE * T_HALF; p += 256) {
    const int row = p >> 5;
    const int tl  = p & 31;
    const int tg  = thalf * T_HALF + tl;  // global tree index

    float P[N_LEAVES];
    P[0] = 1.0f;
#pragma unroll
    for (int l = 0; l < 6; ++l) {
      const int width = 1 << l;
      const int base  = width - 1;        // first node index at level l
#pragma unroll
      for (int i = width - 1; i >= 0; --i) {
        const float g  = sh_g[((base + i) * ROW_TILE + row) * T_HALF + tl];
        const float pl = P[i] * g;
        P[2 * i + 1] = P[i] - pl;         // right child: P * (1 - g)
        P[2 * i]     = pl;                // left  child: P * g
      }
    }

    float acc[LEAF_DIMS];
#pragma unroll
    for (int d = 0; d < LEAF_DIMS; ++d) acc[d] = 0.0f;
#pragma unroll 8
    for (int leaf = 0; leaf < N_LEAVES; ++leaf) {
      const float pr = P[leaf];
#pragma unroll
      for (int d = 0; d < LEAF_DIMS; ++d)
        acc[d] = fmaf(pr, leaf_w[(leaf * LEAF_DIMS + d) * NUM_TREES + tg], acc[d]);
    }
#pragma unroll
    for (int d = 0; d < LEAF_DIMS; ++d)
      sh_red[(row * T_HALF + tl) * LEAF_DIMS + d] = acc[d];
  }
  __syncthreads();

  // -------------------- deterministic per-WG tree reduction ------------------
  float* pp = part + (size_t)thalf * BATCH * LEAF_DIMS;
  for (int o = tid; o < ROW_TILE * LEAF_DIMS; o += 256) {
    const int row = o / LEAF_DIMS;
    const int d   = o - row * LEAF_DIMS;
    float s = 0.0f;
#pragma unroll
    for (int t = 0; t < T_HALF; ++t)
      s += sh_red[(row * T_HALF + t) * LEAF_DIMS + d];
    pp[(size_t)(row0 + row) * LEAF_DIMS + d] = s;
  }
}

// ---------------------------------------------------------------------------
// Kernel C: combine the two tree-half partials (deterministic)
// ---------------------------------------------------------------------------
__global__ __launch_bounds__(256) void sum_parts(
    const float* __restrict__ part, float* __restrict__ out, int n) {
  int i = blockIdx.x * blockDim.x + threadIdx.x;
  if (i < n) out[i] = part[i] + part[n + i];
}

// ---------------------------------------------------------------------------
extern "C" void kernel_launch(void* const* d_in, const int* in_sizes, int n_in,
                              void* d_out, int out_size, void* d_ws, size_t ws_size,
                              hipStream_t stream) {
  const float* x  = (const float*)d_in[0];  // (4096, 1024)
  const float* W  = (const float*)d_in[1];  // (63, 64, 1024)
  const float* b  = (const float*)d_in[2];  // (63, 64)
  const float* lw = (const float*)d_in[3];  // (64, 10, 64)
  float* out = (float*)d_out;               // (4096, 10)

  // workspace layout: xh (f16) | wh (f16) | part (2 x 4096 x 10 f32)
  char* ws = (char*)d_ws;
  _Float16* xh = (_Float16*)ws;
  _Float16* wh = (_Float16*)(ws + (size_t)BATCH * INPUT_DIM * sizeof(_Float16));
  float* part  = (float*)(ws + (size_t)BATCH * INPUT_DIM * sizeof(_Float16)
                             + (size_t)N_INTERNAL * NUM_TREES * INPUT_DIM * sizeof(_Float16));

  const int nx4 = BATCH * INPUT_DIM / 4;
  const int nw4 = N_INTERNAL * NUM_TREES * INPUT_DIM / 4;
  cvt_f32_to_f16_x4<<<(nx4 + 255) / 256, 256, 0, stream>>>(x, xh, nx4);
  cvt_f32_to_f16_x4<<<(nw4 + 255) / 256, 256, 0, stream>>>(W, wh, nw4);

  dim3 grid(BATCH / ROW_TILE, 2);
  const size_t smem =
      (size_t)(N_INTERNAL * ROW_TILE * T_HALF + ROW_TILE * T_HALF * LEAF_DIMS) * sizeof(float);
  soft_tree_main<<<grid, 256, smem, stream>>>(xh, wh, b, lw, part);

  const int n = BATCH * LEAF_DIMS;
  sum_parts<<<(n + 255) / 256, 256, 0, stream>>>(part, out, n);
}
